// minGRU_72945724555721
// MI455X (gfx1250) — compile-verified, matrix operand order checked by reading, not run on hardware
//
#include <hip/hip_runtime.h>
#include <hip/hip_bf16.h>
#include <cstdint>

// ---------------------------------------------------------------------------
// minGRU forward for MI455X (gfx1250, wave32, WMMA).
//   GEMM1: hg = x @ W_hg^T + b_hg       (bf16x3 split-precision WMMA,
//                                        async global->LDS double buffering)
//   scan : log-space linear recurrence  (3-pass chunked associative scan)
//   GEMM2: y  = out @ W_out^T + b_out   (same GEMM kernel)
// ---------------------------------------------------------------------------

typedef __attribute__((ext_vector_type(16))) __bf16 v16bf;
typedef __attribute__((ext_vector_type(8)))  float  v8f;
typedef unsigned short ushort_t;

#define DIM        1024
#define DIM_INNER  1536
#define BATCH      4
#define SEQ        4096
#define MTOT       (BATCH * SEQ)      // 16384 rows for both GEMMs

// GEMM tiling: block 128(M) x 64(N), K-chunk 32, 8 waves as 4(M) x 2(N),
// each wave computes a 32x32 region = 2x2 tiles of 16x16.
#define BM 128
#define BN 64
#define BK 32
#define PD 20          // LDS pitch in dwords for one 32-elem bf16 row (16+4 pad)
// dwords per LDS buffer: Ah, Al tiles (BM rows) + Bh, Bl tiles (BN rows)
#define HALF_DW ((BM * PD) * 2 + (BN * PD) * 2)   // 7680 dwords = 30KB

// Chunked scan: 64 chunks of 64 timesteps.
#define NCH 64
#define CHL 64

// ---------------- scalar math helpers (fp32, matches reference) ------------
__device__ __forceinline__ float softplusf(float x) {
  return fmaxf(x, 0.0f) + log1pf(expf(-fabsf(x)));
}
__device__ __forceinline__ float log_gf(float h) {
  // h>=0 : log(relu(h)+0.5) ; h<0 : -softplus(h) = -log1p(exp(h))
  return (h >= 0.0f) ? logf(h + 0.5f) : -log1pf(expf(h));
}
__device__ __forceinline__ float logaddexpf_(float a, float b) {
  float m = fmaxf(a, b);
  float d = fminf(a, b) - m;              // never both -inf in this kernel
  return m + log1pf(expf(d));
}
// round-to-nearest-even f32 -> bf16 (bit pattern)
__device__ __forceinline__ ushort_t f2bf(float f) {
  uint32_t u = __float_as_uint(f);
  u += 0x7FFFu + ((u >> 16) & 1u);
  return (ushort_t)(u >> 16);
}
__device__ __forceinline__ float bf2f(ushort_t h) {
  return __uint_as_float(((uint32_t)h) << 16);
}

// async global -> LDS, 16 bytes per lane (GV addressing, tracked by ASYNCcnt)
__device__ __forceinline__ void async_b128(uint32_t lds_byte_addr,
                                           const void* gaddr) {
  asm volatile("global_load_async_to_lds_b128 %0, %1, off"
               :
               : "v"(lds_byte_addr), "v"((uint64_t)(uintptr_t)gaddr)
               : "memory");
}
__device__ __forceinline__ void wait_async0() {
  asm volatile("s_wait_asynccnt 0x0" ::: "memory");
}

// ---------------- split fp32 -> (hi, lo) bf16 -------------------------------
__global__ __launch_bounds__(256) void split_bf16x2(
    const float* __restrict__ src, ushort_t* __restrict__ hi,
    ushort_t* __restrict__ lo, int n) {
  int i = blockIdx.x * blockDim.x + threadIdx.x;
  if (i < n) {
    float x = src[i];
    ushort_t h = f2bf(x);
    hi[i] = h;
    lo[i] = f2bf(x - bf2f(h));
  }
}

// ---------------- bf16x3 split-precision GEMM -------------------------------
// C[m][n] = sum_k A[m][k]*B[n][k] + bias[n]
// A given as hi/lo bf16 [M x K] row-major, B as hi/lo bf16 [N x K] row-major.
__global__ __launch_bounds__(256) void gemm_bf16x3(
    const ushort_t* __restrict__ Ah, const ushort_t* __restrict__ Al,
    const ushort_t* __restrict__ Bh, const ushort_t* __restrict__ Bl,
    const float* __restrict__ bias, float* __restrict__ C,
    int M, int N, int K) {
  __shared__ __align__(16) uint32_t lds[2 * HALF_DW];   // double buffered

  const int tid  = threadIdx.x;
  const int lane = tid & 31;
  const int wave = tid >> 5;      // 0..7
  const int wm   = wave & 3;      // wave row (M)
  const int wn   = wave >> 2;     // wave col (N)
  const int block_m = blockIdx.x * BM;
  const int block_n = blockIdx.y * BN;

  const uint32_t lds_base = (uint32_t)(uintptr_t)&lds[0];  // LDS byte address

  // ISA 16-bit A/B fragment layout (wave32):
  //   lanes 0-15 : M(or N)=lane,    VGPR0..3 -> K 0..7,  VGPR4..7 -> K 16..23
  //   lanes16-31 : M(or N)=lane-16, VGPR0..3 -> K 8..15, VGPR4..7 -> K 24..31
  const int frag_row = lane & 15;
  const int kb_dw    = (lane < 16) ? 0 : 4;   // dword offset of K-base in row

  v8f acc[2][2] = {};
  union Frag { v16bf v; uint4 q[2]; };

  // issue async loads for one K-chunk into LDS buffer `buf` (uniform EXEC)
  auto issue_chunk = [&](int kc, int buf) {
    const uint32_t base = lds_base + (uint32_t)buf * (HALF_DW * 4u);
#pragma unroll
    for (int it = 0; it < 2; ++it) {           // A tile: 128 rows x 32 bf16
      int u = tid + it * 256;                  // 512 uint4 units, 2/thread
      int row = u >> 2, seg = u & 3;
      uint32_t d = base + (uint32_t)(row * PD + seg * 4) * 4u;
      async_b128(d, Ah + (size_t)(block_m + row) * K + kc + seg * 8);
      async_b128(d + (uint32_t)(BM * PD * 4),
                 Al + (size_t)(block_m + row) * K + kc + seg * 8);
    }
    {                                          // B tile: 64 rows x 32 bf16
      int row = tid >> 2, seg = tid & 3;       // 256 units, 1/thread
      uint32_t d = base + (uint32_t)(2 * BM * PD * 4) +
                   (uint32_t)(row * PD + seg * 4) * 4u;
      async_b128(d, Bh + (size_t)(block_n + row) * K + kc + seg * 8);
      async_b128(d + (uint32_t)(BN * PD * 4),
                 Bl + (size_t)(block_n + row) * K + kc + seg * 8);
    }
  };

  const int nck = K / BK;
  issue_chunk(0, 0);

  for (int i = 0; i < nck; ++i) {
    wait_async0();        // my async loads for chunk i have landed in LDS
    __syncthreads();      // all waves' loads visible; prev buffer reads done
    if (i + 1 < nck) issue_chunk((i + 1) * BK, (i + 1) & 1);

    const uint32_t bofs = (uint32_t)(i & 1) * HALF_DW;  // dword offset
    const uint32_t* sAh = lds + bofs;
    const uint32_t* sAl = sAh + BM * PD;
    const uint32_t* sBh = sAl + BM * PD;
    const uint32_t* sBl = sBh + BN * PD;

    // ---- LDS -> fragments (2 x ds_load_b128 each) ----
    Frag fa_h[2], fa_l[2], fb_h[2], fb_l[2];
#pragma unroll
    for (int t = 0; t < 2; ++t) {
      int ar = (wm * 32 + t * 16 + frag_row) * PD + kb_dw;
      fa_h[t].q[0] = *(const uint4*)&sAh[ar];
      fa_h[t].q[1] = *(const uint4*)&sAh[ar + 8];
      fa_l[t].q[0] = *(const uint4*)&sAl[ar];
      fa_l[t].q[1] = *(const uint4*)&sAl[ar + 8];
      int br = (wn * 32 + t * 16 + frag_row) * PD + kb_dw;
      fb_h[t].q[0] = *(const uint4*)&sBh[br];
      fb_h[t].q[1] = *(const uint4*)&sBh[br + 8];
      fb_l[t].q[0] = *(const uint4*)&sBl[br];
      fb_l[t].q[1] = *(const uint4*)&sBl[br + 8];
    }

    // ---- 12 x v_wmma_f32_16x16x32_bf16 (hi*hi + hi*lo + lo*hi) ----
#pragma unroll
    for (int ii = 0; ii < 2; ++ii) {
#pragma unroll
      for (int jj = 0; jj < 2; ++jj) {
        acc[ii][jj] = __builtin_amdgcn_wmma_f32_16x16x32_bf16(
            false, fa_h[ii].v, false, fb_h[jj].v, (short)0, acc[ii][jj], false,
            false);
        acc[ii][jj] = __builtin_amdgcn_wmma_f32_16x16x32_bf16(
            false, fa_h[ii].v, false, fb_l[jj].v, (short)0, acc[ii][jj], false,
            false);
        acc[ii][jj] = __builtin_amdgcn_wmma_f32_16x16x32_bf16(
            false, fa_l[ii].v, false, fb_h[jj].v, (short)0, acc[ii][jj], false,
            false);
      }
    }
  }

  // ---- epilogue: C/D layout -> global (row-coalesced dword stores) ----
  const int col0  = lane & 15;
  const int rbase = (lane < 16) ? 0 : 8;
#pragma unroll
  for (int i = 0; i < 2; ++i) {
#pragma unroll
    for (int j = 0; j < 2; ++j) {
      int gm0 = block_m + wm * 32 + i * 16 + rbase;
      int gn  = block_n + wn * 32 + j * 16 + col0;
      float bv = bias[gn];
#pragma unroll
      for (int r = 0; r < 8; ++r) {
        C[(size_t)(gm0 + r) * N + gn] = acc[i][j][r] + bv;
      }
    }
  }
}

// ---------------- scan pass 1: per-chunk local scan + aggregates ------------
__global__ __launch_bounds__(256) void scan_pass1(
    const float* __restrict__ hg, float* __restrict__ Asum,
    float* __restrict__ Vend) {
  int tid = blockIdx.x * blockDim.x + threadIdx.x;   // BATCH*NCH*DIM_INNER
  int c = tid % DIM_INNER;
  int j = (tid / DIM_INNER) % NCH;
  int b = tid / (DIM_INNER * NCH);
  const float* base = hg + (size_t)(b * SEQ + j * CHL) * (2 * DIM_INNER);
  float A = 0.0f, V = -INFINITY;
  for (int s = 0; s < CHL; ++s) {
    float h  = base[(size_t)s * (2 * DIM_INNER) + c];
    float g  = base[(size_t)s * (2 * DIM_INNER) + DIM_INNER + c];
    float sp = softplusf(g);
    float lc = -sp;                       // log_coeffs
    float lv = (g - sp) + log_gf(h);      // log_z + log_g(hidden)
    A += lc;
    V = logaddexpf_(V + lc, lv);
  }
  size_t o = ((size_t)b * NCH + j) * DIM_INNER + c;
  Asum[o] = A;
  Vend[o] = V;
}

// ---------------- scan pass 2: carry scan across chunks ---------------------
__global__ __launch_bounds__(256) void scan_pass2(
    const float* __restrict__ Asum, const float* __restrict__ Vend,
    const float* __restrict__ h0, float* __restrict__ enter) {
  int tid = blockIdx.x * blockDim.x + threadIdx.x;   // BATCH*DIM_INNER
  if (tid >= BATCH * DIM_INNER) return;
  int c = tid % DIM_INNER;
  int b = tid / DIM_INNER;
  float st = log_gf(h0[c]);
  for (int j = 0; j < NCH; ++j) {
    size_t o = ((size_t)b * NCH + j) * DIM_INNER + c;
    enter[o] = st;
    st = logaddexpf_(st + Asum[o], Vend[o]);
  }
}

// ---------------- scan pass 3: final states -> out (bf16 hi/lo) -------------
__global__ __launch_bounds__(256) void scan_pass3(
    const float* __restrict__ hg, const float* __restrict__ enter,
    ushort_t* __restrict__ outh, ushort_t* __restrict__ outl,
    float* __restrict__ nh, float* __restrict__ nlh) {
  int tid = blockIdx.x * blockDim.x + threadIdx.x;
  int c = tid % DIM_INNER;
  int j = (tid / DIM_INNER) % NCH;
  int b = tid / (DIM_INNER * NCH);
  const float* base = hg + (size_t)(b * SEQ + j * CHL) * (2 * DIM_INNER);
  float st = enter[((size_t)b * NCH + j) * DIM_INNER + c];
  for (int s = 0; s < CHL; ++s) {
    float h  = base[(size_t)s * (2 * DIM_INNER) + c];
    float g  = base[(size_t)s * (2 * DIM_INNER) + DIM_INNER + c];
    float sp = softplusf(g);
    st = logaddexpf_(st - sp, (g - sp) + log_gf(h));
    float o = expf(st);
    size_t oi = (size_t)(b * SEQ + j * CHL + s) * DIM_INNER + c;
    ushort_t oh = f2bf(o);
    outh[oi] = oh;
    outl[oi] = f2bf(o - bf2f(oh));
    if (j == NCH - 1 && s == CHL - 1) {
      nh[b * DIM_INNER + c]  = o;    // next_hidden
      nlh[b * DIM_INNER + c] = st;   // next_log_hidden
    }
  }
}

// ---------------------------------------------------------------------------
extern "C" void kernel_launch(void* const* d_in, const int* in_sizes, int n_in,
                              void* d_out, int out_size, void* d_ws,
                              size_t ws_size, hipStream_t stream) {
  (void)in_sizes; (void)n_in; (void)out_size; (void)ws_size;
  const float* x     = (const float*)d_in[0];   // [B,S,DIM]
  const float* W_hg  = (const float*)d_in[1];   // [2*DIM_INNER, DIM]
  const float* b_hg  = (const float*)d_in[2];   // [2*DIM_INNER]
  const float* W_out = (const float*)d_in[3];   // [DIM, DIM_INNER]
  const float* b_out = (const float*)d_in[4];   // [DIM]
  const float* h0    = (const float*)d_in[5];   // [DIM_INNER]

  // carve workspace (256B aligned slices)
  char* p = (char*)d_ws;
  auto carve = [&](size_t bytes) {
    char* r = p;
    p += (bytes + 255) & ~(size_t)255;
    return r;
  };
  const size_t nX   = (size_t)MTOT * DIM;            // 16.7M
  const size_t nWhg = (size_t)(2 * DIM_INNER) * DIM; // 3.1M
  const size_t nWo  = (size_t)DIM * DIM_INNER;       // 1.6M
  const size_t nOut = (size_t)MTOT * DIM_INNER;      // 25.2M
  ushort_t* xh   = (ushort_t*)carve(nX * 2);
  ushort_t* xl   = (ushort_t*)carve(nX * 2);
  ushort_t* whgh = (ushort_t*)carve(nWhg * 2);
  ushort_t* whgl = (ushort_t*)carve(nWhg * 2);
  ushort_t* woh  = (ushort_t*)carve(nWo * 2);
  ushort_t* wol  = (ushort_t*)carve(nWo * 2);
  float*    hg   = (float*)carve((size_t)MTOT * 2 * DIM_INNER * 4); // 192MB
  ushort_t* outh = (ushort_t*)carve(nOut * 2);
  ushort_t* outl = (ushort_t*)carve(nOut * 2);
  float*    Asum = (float*)carve((size_t)BATCH * NCH * DIM_INNER * 4);
  float*    Vend = (float*)carve((size_t)BATCH * NCH * DIM_INNER * 4);
  float*    entr = (float*)carve((size_t)BATCH * NCH * DIM_INNER * 4);

  float* y   = (float*)d_out;                        // [B,S,DIM]
  float* nh  = y + (size_t)MTOT * DIM;               // [B,1,DIM_INNER]
  float* nlh = nh + (size_t)BATCH * DIM_INNER;       // [B,1,DIM_INNER]

  // 1) split fp32 inputs into bf16 hi/lo
  split_bf16x2<<<(int)((nX + 255) / 256), 256, 0, stream>>>(x, xh, xl, (int)nX);
  split_bf16x2<<<(int)((nWhg + 255) / 256), 256, 0, stream>>>(W_hg, whgh, whgl,
                                                              (int)nWhg);
  split_bf16x2<<<(int)((nWo + 255) / 256), 256, 0, stream>>>(W_out, woh, wol,
                                                             (int)nWo);

  // 2) GEMM1: hg = x @ W_hg^T + b_hg   (M=16384, N=3072, K=1024)
  gemm_bf16x3<<<dim3(MTOT / BM, (2 * DIM_INNER) / BN), 256, 0, stream>>>(
      xh, xl, whgh, whgl, b_hg, hg, MTOT, 2 * DIM_INNER, DIM);

  // 3) log-space associative scan (3 passes)
  const int scan_threads = BATCH * NCH * DIM_INNER;  // 393216
  scan_pass1<<<scan_threads / 256, 256, 0, stream>>>(hg, Asum, Vend);
  scan_pass2<<<(BATCH * DIM_INNER + 255) / 256, 256, 0, stream>>>(Asum, Vend,
                                                                  h0, entr);
  scan_pass3<<<scan_threads / 256, 256, 0, stream>>>(hg, entr, outh, outl, nh,
                                                     nlh);

  // 4) GEMM2: y = out @ W_out^T + b_out (M=16384, N=1024, K=1536)
  gemm_bf16x3<<<dim3(MTOT / BM, DIM / BN), 256, 0, stream>>>(
      outh, outl, woh, wol, b_out, y, MTOT, DIM, DIM_INNER);
}
